// GAT_56727928046275
// MI455X (gfx1250) — matrix operand order repaired and checked by reading, not compiled, not run on hardware
//
#include <hip/hip_runtime.h>
#include <math.h>

// ---------------------------------------------------------------------------
// Types for CDNA5 WMMA
// ---------------------------------------------------------------------------
typedef __attribute__((ext_vector_type(16))) _Float16 v16h;
typedef __attribute__((ext_vector_type(8)))  _Float16 v8h;
typedef __attribute__((ext_vector_type(4)))  _Float16 v4h;
typedef __attribute__((ext_vector_type(8)))  float    v8f;

#define TM 64
#define TN 64
#define TK 32

// ---------------------------------------------------------------------------
// WMMA GEMM: C[M,N] = A[M,K] * B[K,N] + bias[N]   (f32 in/out, f16 compute)
// Block: 128 threads = 4 waves. Wave w computes rows [w*16, w*16+16) of a
// 64x64 block tile; 4 accumulators of 16x16 each across the N dimension.
// K, N multiples of 32 / 64 (true here: K in {128,192}, N in {192,64}).
// M handled by clamping staged rows (pad rows only feed unstored output
// rows) and guarding the final store.
// ---------------------------------------------------------------------------
__global__ __launch_bounds__(128) void wmma_gemm_bias(
    const float* __restrict__ A, const float* __restrict__ B,
    const float* __restrict__ bias, float* __restrict__ C,
    int M, int K, int N)
{
    __shared__ __align__(64) _Float16 sA[TM * TK];   // row-major [64][32]
    __shared__ __align__(64) _Float16 sB[TN * TK];   // transposed: [n][k] = [64][32]

    const int t    = threadIdx.x;
    const int lane = t & 31;
    const int wave = t >> 5;
    const int m0   = blockIdx.x * TM;
    const int n0   = blockIdx.y * TN;
    const int hi   = (lane >> 4) & 1;
    const int l16  = lane & 15;

    v8f acc[4];
    v8f zero = {};
#pragma unroll
    for (int i = 0; i < 4; ++i) acc[i] = zero;

    for (int kc = 0; kc < K; kc += TK) {
        // ---- Stage A tile (64 rows x 32 cols): vectorized float4 -> v4h ----
        // Row index clamped instead of guarded: out-of-range rows only feed
        // accumulator rows that are never stored, so no branch, no exec mask,
        // and all 4 b128 loads can issue back-to-back under one wait.
#pragma unroll
        for (int j = 0; j < 4; ++j) {
            int f  = t + 128 * j;      // float4 slot 0..511
            int r  = f >> 3;           // row 0..63
            int c4 = f & 7;            // col quad 0..7
            int gr = m0 + r;
            gr = (gr < M) ? gr : (M - 1);
            float4 v = *(const float4*)&A[(size_t)gr * K + kc + c4 * 4];
            v4h hv = { (_Float16)v.x, (_Float16)v.y, (_Float16)v.z, (_Float16)v.w };
            *(v4h*)&sA[r * TK + c4 * 4] = hv;
        }
        // ---- Stage B tile (32 k-rows x 64 n-cols) transposed into [n][k] ----
        // Each thread handles a (column nn, k-quad kq): 4 lane-coalesced b32
        // loads along K, one contiguous ds_store_b64 into the [n][k] layout.
#pragma unroll
        for (int j = 0; j < 4; ++j) {
            int f  = t + 128 * j;      // task 0..511
            int nn = f & 63;           // column 0..63
            int kq = f >> 6;           // k quad 0..7
            const float* bp = &B[(size_t)(kc + kq * 4) * N + n0 + nn];
            float b0 = bp[0];
            float b1 = bp[(size_t)N];
            float b2 = bp[(size_t)2 * N];
            float b3 = bp[(size_t)3 * N];
            v4h hv = { (_Float16)b0, (_Float16)b1, (_Float16)b2, (_Float16)b3 };
            *(v4h*)&sB[nn * TK + kq * 4] = hv;
        }
        __syncthreads();

        // A fragment: lane l16 (+hi half) holds row (wave*16+l16),
        // K = hi*16 .. hi*16+15 (contiguous, matching B fragment)
        const _Float16* pa = &sA[(wave * 16 + l16) * TK + hi * 16];
        v8h alo = *(const v8h*)(pa);
        v8h ahi = *(const v8h*)(pa + 8);
        v16h afrag = __builtin_shufflevector(alo, ahi,
            0, 1, 2, 3, 4, 5, 6, 7, 8, 9, 10, 11, 12, 13, 14, 15);

#pragma unroll
        for (int nt = 0; nt < 4; ++nt) {
            // B fragment: lane l16 holds column (nt*16+l16), K = hi*16..hi*16+15,
            // contiguous because sB is stored [n][k].
            const _Float16* pb = &sB[(nt * 16 + l16) * TK + hi * 16];
            v16h bfrag = *(const v16h*)pb;
            acc[nt] = __builtin_amdgcn_wmma_f32_16x16x32_f16(
                false, afrag, false, bfrag, (short)0, acc[nt], false, false);
        }
        __syncthreads();
    }

    // Store: C/D layout — VGPR r: lanes 0-15 -> M=r, lanes 16-31 -> M=r+8.
#pragma unroll
    for (int nt = 0; nt < 4; ++nt) {
        int col = n0 + nt * 16 + l16;
        float bv = bias[col];
#pragma unroll
        for (int r = 0; r < 8; ++r) {
            int row = m0 + wave * 16 + hi * 8 + r;
            if (row < M) C[(size_t)row * N + col] = acc[nt][r] + bv;
        }
    }
}

// ---------------------------------------------------------------------------
// Monotonic float <-> u32 key for atomicMax-based segment max
// ---------------------------------------------------------------------------
__device__ __forceinline__ unsigned fkey(float f) {
    unsigned u = __float_as_uint(f);
    return (u & 0x80000000u) ? ~u : (u | 0x80000000u);
}
__device__ __forceinline__ float funkey(unsigned k) {
    unsigned u = (k & 0x80000000u) ? (k & 0x7fffffffu) : ~k;
    return __uint_as_float(u);
}

// ---------------------------------------------------------------------------
// Init helpers (graph-capture safe; no hipMemset)
// ---------------------------------------------------------------------------
__global__ void fill_u32(unsigned* __restrict__ p, unsigned v, int n) {
    int i = blockIdx.x * blockDim.x + threadIdx.x;
    if (i < n) p[i] = v;
}
__global__ void init_bias(float* __restrict__ o, const float* __restrict__ b,
                          int n, int C) {
    int i = blockIdx.x * blockDim.x + threadIdx.x;
    if (i < n * C) o[i] = b[i % C];
}

// ---------------------------------------------------------------------------
// Edge pass 1: logit[e,h] = sum_c leaky(xl[src]+xr[dst]) * att ; atomicMax per dst
// One wave32 per edge. C == 64 fixed, H templated (3 or 1).
// ---------------------------------------------------------------------------
template <int H>
__global__ __launch_bounds__(256) void gat_edge_logits(
    const float* __restrict__ xl, const float* __restrict__ xr,
    const float* __restrict__ att, const int* __restrict__ ei,
    unsigned* __restrict__ maxkey, float* __restrict__ logit,
    int nEdges, int totEdges)
{
    const int lane = threadIdx.x & 31;
    const int e = blockIdx.x * (blockDim.x >> 5) + (threadIdx.x >> 5);
    if (e >= totEdges) return;

    int src, dst;
    if (e < nEdges) { src = ei[e]; dst = ei[nEdges + e]; }
    else            { src = e - nEdges; dst = src; }

    const int HC = H * 64;
    const float* pl = xl + (size_t)src * HC;
    const float* pr = xr + (size_t)dst * HC;

    float s[H];
#pragma unroll
    for (int h = 0; h < H; ++h) s[h] = 0.0f;

#pragma unroll
    for (int i = 0; i < 2 * H; ++i) {
        int c = lane + 32 * i;
        float v = pl[c] + pr[c];
        v = (v > 0.0f) ? v : 0.2f * v;          // leaky_relu(0.2)
        s[i >> 1] += v * att[c];
    }
#pragma unroll
    for (int h = 0; h < H; ++h) {
#pragma unroll
        for (int off = 16; off > 0; off >>= 1)
            s[h] += __shfl_xor(s[h], off, 32);
    }
    if (lane < H) {
        float lg = s[0];
        if (H > 1 && lane == 1) lg = s[(H > 1) ? 1 : 0];
        if (H > 2 && lane == 2) lg = s[(H > 2) ? 2 : 0];
        logit[(size_t)e * H + lane] = lg;
        atomicMax(&maxkey[(size_t)dst * H + lane], fkey(lg));
    }
}

// ---------------------------------------------------------------------------
// Edge pass 2: p = exp(logit - max[dst]); denom[dst] += p (in-place overwrite)
// ---------------------------------------------------------------------------
__global__ void gat_edge_softmax_denom(
    float* __restrict__ logit, const unsigned* __restrict__ maxkey,
    float* __restrict__ denom, const int* __restrict__ ei,
    int nEdges, int totEdges, int H)
{
    int tid = blockIdx.x * blockDim.x + threadIdx.x;
    int tot = totEdges * H;
    if (tid >= tot) return;
    int e = tid / H;
    int h = tid - e * H;
    int dst = (e < nEdges) ? ei[nEdges + e] : (e - nEdges);
    float m = funkey(maxkey[(size_t)dst * H + h]);
    float p = __expf(logit[tid] - m);
    logit[tid] = p;
    atomicAdd(&denom[(size_t)dst * H + h], p);
}

// ---------------------------------------------------------------------------
// Edge pass 3: out[dst] += (p/denom[dst]) * xl[src]    (one wave32 per edge)
// ---------------------------------------------------------------------------
template <int H>
__global__ __launch_bounds__(256) void gat_edge_aggregate(
    const float* __restrict__ xl, const float* __restrict__ p,
    const float* __restrict__ denom, const int* __restrict__ ei,
    float* __restrict__ out, int nEdges, int totEdges)
{
    const int lane = threadIdx.x & 31;
    const int e = blockIdx.x * (blockDim.x >> 5) + (threadIdx.x >> 5);
    if (e >= totEdges) return;

    int src, dst;
    if (e < nEdges) { src = ei[e]; dst = ei[nEdges + e]; }
    else            { src = e - nEdges; dst = src; }

    const int HC = H * 64;
    float alpha[H];
#pragma unroll
    for (int h = 0; h < H; ++h)
        alpha[h] = p[(size_t)e * H + h] / denom[(size_t)dst * H + h];

    const float* pl = xl + (size_t)src * HC;
    float* po = out + (size_t)dst * HC;
#pragma unroll
    for (int i = 0; i < 2 * H; ++i) {
        int c = lane + 32 * i;
        atomicAdd(&po[c], alpha[i >> 1] * pl[c]);
    }
}

// ---------------------------------------------------------------------------
// BatchNorm over nodes: relu applied to input; two-pass (partial sums, norm)
// sums[0..C) = sum(relu(h)), sums[C..2C) = sum(relu(h)^2)
// ---------------------------------------------------------------------------
__global__ void bn_partial(const float* __restrict__ h, float* __restrict__ sums,
                           int n, int C)
{
    int ch = threadIdx.x;                       // blockDim.x == C
    float s = 0.0f, q = 0.0f;
    for (int r = blockIdx.x; r < n; r += gridDim.x) {
        float v = h[(size_t)r * C + ch];
        v = (v > 0.0f) ? v : 0.0f;
        s += v;
        q += v * v;
    }
    atomicAdd(&sums[ch], s);
    atomicAdd(&sums[C + ch], q);
}

__global__ void bn_norm(const float* __restrict__ h, const float* __restrict__ sums,
                        const float* __restrict__ gamma, const float* __restrict__ beta,
                        float* __restrict__ y, int n, int C)
{
    int tid = blockIdx.x * blockDim.x + threadIdx.x;
    if (tid >= n * C) return;
    int ch = tid % C;
    float invn = 1.0f / (float)n;
    float mu = sums[ch] * invn;
    float var = sums[C + ch] * invn - mu * mu;
    float v = h[tid];
    v = (v > 0.0f) ? v : 0.0f;
    y[tid] = (v - mu) * rsqrtf(var + 1e-5f) * gamma[ch] + beta[ch];
}

// ---------------------------------------------------------------------------
// Host-side orchestration
// ---------------------------------------------------------------------------
extern "C" void kernel_launch(void* const* d_in, const int* in_sizes, int n_in,
                              void* d_out, int out_size, void* d_ws, size_t ws_size,
                              hipStream_t stream)
{
    (void)n_in; (void)out_size; (void)ws_size;

    const float* x    = (const float*)d_in[0];
    const int*   ei   = (const int*)  d_in[1];
    const float* Wl0  = (const float*)d_in[2];
    const float* bl0  = (const float*)d_in[3];
    const float* Wr0  = (const float*)d_in[4];
    const float* br0  = (const float*)d_in[5];
    const float* att0 = (const float*)d_in[6];
    const float* bias0= (const float*)d_in[7];
    const float* g0   = (const float*)d_in[8];
    const float* be0  = (const float*)d_in[9];
    const float* Wl1  = (const float*)d_in[10];
    const float* bl1  = (const float*)d_in[11];
    const float* Wr1  = (const float*)d_in[12];
    const float* br1  = (const float*)d_in[13];
    const float* att1 = (const float*)d_in[14];
    const float* bias1= (const float*)d_in[15];
    const float* g1   = (const float*)d_in[16];
    const float* be1  = (const float*)d_in[17];

    const int IN   = 128;
    const int Nn   = in_sizes[0] / IN;     // 50000
    const int E    = in_sizes[1] / 2;      // 800000
    const int ET   = E + Nn;               // with self loops
    const int HC0  = 192;                  // heads*hid layer 0
    const int C1   = 64;                   // layer 1 channels

    // ---- workspace carve-out (256B aligned) ----
    char* cur = (char*)d_ws;
    auto carve = [&](size_t bytes) -> void* {
        void* r = (void*)cur;
        cur += (bytes + 255) & ~(size_t)255;
        return r;
    };
    float*    xl0   = (float*)   carve((size_t)Nn * HC0 * 4);
    float*    xr0   = (float*)   carve((size_t)Nn * HC0 * 4);   // reused as y0
    float*    h0    = (float*)   carve((size_t)Nn * HC0 * 4);
    float*    xl1   = (float*)   carve((size_t)Nn * C1 * 4);
    float*    xr1   = (float*)   carve((size_t)Nn * C1 * 4);
    float*    h1    = (float*)   carve((size_t)Nn * C1 * 4);
    float*    lg0   = (float*)   carve((size_t)ET * 3 * 4);
    float*    lg1   = (float*)   carve((size_t)ET * 1 * 4);
    unsigned* mk0   = (unsigned*)carve((size_t)Nn * 3 * 4);
    unsigned* mk1   = (unsigned*)carve((size_t)Nn * 1 * 4);
    float*    den0  = (float*)   carve((size_t)Nn * 3 * 4);
    float*    den1  = (float*)   carve((size_t)Nn * 1 * 4);
    float*    sums  = (float*)   carve((size_t)2 * HC0 * 4);

    const unsigned KEY_NEG_INF = 0x007FFFFFu;  // fkey(-inf)

    const int mBlocks = (Nn + TM - 1) / TM;
    const int edgeBlocks = (ET + 7) / 8;       // 8 waves/block, 1 wave/edge

    // ================= Layer 0 =================
    {
        dim3 g(mBlocks, HC0 / TN);
        wmma_gemm_bias<<<g, 128, 0, stream>>>(x, Wl0, bl0, xl0, Nn, IN, HC0);
        wmma_gemm_bias<<<g, 128, 0, stream>>>(x, Wr0, br0, xr0, Nn, IN, HC0);
    }
    fill_u32<<<(Nn * 3 + 255) / 256, 256, 0, stream>>>(mk0, KEY_NEG_INF, Nn * 3);
    fill_u32<<<(Nn * 3 + 255) / 256, 256, 0, stream>>>((unsigned*)den0, 0u, Nn * 3);
    init_bias<<<((Nn * HC0) + 255) / 256, 256, 0, stream>>>(h0, bias0, Nn, HC0);

    gat_edge_logits<3><<<edgeBlocks, 256, 0, stream>>>(xl0, xr0, att0, ei, mk0, lg0, E, ET);
    gat_edge_softmax_denom<<<(ET * 3 + 255) / 256, 256, 0, stream>>>(lg0, mk0, den0, ei, E, ET, 3);
    gat_edge_aggregate<3><<<edgeBlocks, 256, 0, stream>>>(xl0, lg0, den0, ei, h0, E, ET);

    fill_u32<<<(2 * HC0 + 255) / 256, 256, 0, stream>>>((unsigned*)sums, 0u, 2 * HC0);
    bn_partial<<<512, HC0, 0, stream>>>(h0, sums, Nn, HC0);
    float* y0 = xr0;  // reuse
    bn_norm<<<((Nn * HC0) + 255) / 256, 256, 0, stream>>>(h0, sums, g0, be0, y0, Nn, HC0);

    // ================= Layer 1 =================
    {
        dim3 g(mBlocks, C1 / TN);
        wmma_gemm_bias<<<g, 128, 0, stream>>>(y0, Wl1, bl1, xl1, Nn, HC0, C1);
        wmma_gemm_bias<<<g, 128, 0, stream>>>(y0, Wr1, br1, xr1, Nn, HC0, C1);
    }
    fill_u32<<<(Nn + 255) / 256, 256, 0, stream>>>(mk1, KEY_NEG_INF, Nn);
    fill_u32<<<(Nn + 255) / 256, 256, 0, stream>>>((unsigned*)den1, 0u, Nn);
    init_bias<<<((Nn * C1) + 255) / 256, 256, 0, stream>>>(h1, bias1, Nn, C1);

    gat_edge_logits<1><<<edgeBlocks, 256, 0, stream>>>(xl1, xr1, att1, ei, mk1, lg1, E, ET);
    gat_edge_softmax_denom<<<(ET + 255) / 256, 256, 0, stream>>>(lg1, mk1, den1, ei, E, ET, 1);
    gat_edge_aggregate<1><<<edgeBlocks, 256, 0, stream>>>(xl1, lg1, den1, ei, h1, E, ET);

    fill_u32<<<(2 * C1 + 255) / 256, 256, 0, stream>>>((unsigned*)sums, 0u, 2 * C1);
    bn_partial<<<512, C1, 0, stream>>>(h1, sums, Nn, C1);
    bn_norm<<<((Nn * C1) + 255) / 256, 256, 0, stream>>>(h1, sums, g1, be1, (float*)d_out, Nn, C1);
}
